// EpiNN_aaindex_24807731101786
// MI455X (gfx1250) — matrix operand order repaired and checked by reading, not compiled, not run on hardware
//
#include <hip/hip_runtime.h>

#define LL 256
#define DD 64
#define BB 32

typedef __attribute__((ext_vector_type(16))) _Float16 v16h;
typedef __attribute__((ext_vector_type(8)))  _Float16 v8h;
typedef __attribute__((ext_vector_type(8)))  float    v8f;
typedef __attribute__((ext_vector_type(8)))  unsigned int v8u;

// ---------------- workspace layout (bytes) ----------------
static const size_t OFF_X2W = 0;                       // 32*256*64 f16  = 1,048,576
static const size_t OFF_UH  = 1048576;                 // 32*256*64 f32  = 2,097,152
static const size_t OFF_MSK = OFF_UH  + 2097152;       // 32*256   f32   = 32,768
static const size_t OFF_W1A = OFF_MSK + 32768;         // 64*64    f16   = 8,192
static const size_t OFF_W2H = OFF_W1A + 8192;          // 16*64    f16   = 2,048
static const size_t OFF_PAR = OFF_W2H + 2048;          // 32*16*16 f32   = 32,768

// Load a v16h WMMA fragment from two contiguous 16-byte chunks.
__device__ __forceinline__ v16h ld_frag2(const _Float16* p0, const _Float16* p1) {
  v8h a = *(const v8h*)p0;
  v8h b = *(const v8h*)p1;
  v16h r;
#pragma unroll
  for (int t = 0; t < 8; ++t) { r[t] = a[t]; r[t + 8] = b[t]; }
  return r;
}

// |a-b| elementwise f16: packed sub then clear sign bits (v_pk_add + v_and).
__device__ __forceinline__ v16h abs_diff(v16h a, v16h b) {
  union { v16h h; v8u u; } x;
  x.h = a - b;
#pragma unroll
  for (int t = 0; t < 8; ++t) x.u[t] &= 0x7FFF7FFFu;
  return x.h;
}

// leaky relu as a single max: for v>0, v > 0.01v; for v<0, 0.01v > v.
__device__ __forceinline__ float lrelu(float v) { return fmaxf(v, 0.01f * v); }

// ---------------- prep: x1 -> out, x2w (f16), mask ----------------
__global__ __launch_bounds__(256)
void prep_kernel(const float* __restrict__ emb, const float* __restrict__ tw,
                 const float* __restrict__ tb, _Float16* __restrict__ x2w,
                 float* __restrict__ maskA, float* __restrict__ out) {
  const int b = blockIdx.x;
  const int t = threadIdx.x;            // t == token index l
  const float* e = emb + (size_t)b * (LL * DD + 1);
  float asum = 0.f, xsum = 0.f;
#pragma unroll 4
  for (int d = 0; d < DD; ++d) {
    float ev  = e[t * DD + d];
    float twv = tw[t * DD + d];
    float p   = ev * twv;
    x2w[((size_t)b * LL + t) * DD + d] = (_Float16)p;
    asum += fabsf(ev);
    xsum += p;
  }
  maskA[b * LL + t] = (asum > 0.f) ? 1.f : 0.f;
  __shared__ float red[256];
  red[t] = xsum;
  __syncthreads();
  for (int s = 128; s > 0; s >>= 1) { if (t < s) red[t] += red[t + s]; __syncthreads(); }
  if (t == 0) out[b] = red[0] + e[LL * DD] * tw[LL * DD] + tb[0];
}

// ---------------- wprep: f16 weight copies (B-layout friendly, n-major) ---
__global__ __launch_bounds__(256)
void wprep_kernel(const float* __restrict__ w1, const float* __restrict__ w2,
                  _Float16* __restrict__ w1a, _Float16* __restrict__ w2h) {
  int t = blockIdx.x * 256 + threadIdx.x;
  if (t < 4096) {                       // w1a[d][dp] = w1[d][64+dp]
    int d = t >> 6, dp = t & 63;
    w1a[d * DD + dp] = (_Float16)w1[d * 128 + 64 + dp];
  }
  if (t < 1024) w2h[t] = (_Float16)w2[t];   // w2h[k2][d] row-major copy
}

// ---------------- uh[b,i,d] = 0.5 * sum_dp x2[b,i,dp]*w1[d,dp] -----------
__global__ __launch_bounds__(64)
void uh_kernel(const float* __restrict__ emb, const float* __restrict__ tw,
               const float* __restrict__ w1, float* __restrict__ uh) {
  const int row = blockIdx.x;           // b*L + i
  const int b = row / LL, i = row % LL;
  const int d = threadIdx.x;
  __shared__ float sx[DD];
  const float* e = emb + (size_t)b * (LL * DD + 1) + i * DD;
  sx[d] = e[d] * tw[i * DD + d];
  __syncthreads();
  float acc = 0.f;
#pragma unroll 8
  for (int dp = 0; dp < DD; ++dp) acc += sx[dp] * w1[d * 128 + dp];
  uh[(size_t)row * DD + d] = 0.5f * acc;
}

// ---------------- main pairwise WMMA kernel ------------------------------
__global__ __launch_bounds__(128)
void pair_kernel(const _Float16* __restrict__ x2w, const float* __restrict__ uh,
                 const float* __restrict__ maskA,
                 const _Float16* __restrict__ w1a, const _Float16* __restrict__ w2h,
                 const float* __restrict__ b1, const float* __restrict__ b2,
                 const float* __restrict__ w3, const float* __restrict__ b3,
                 float* __restrict__ partial) {
  const int b   = blockIdx.z;
  const int It  = blockIdx.x;           // i-tile (16 rows)
  const int Jg  = blockIdx.y;           // j-group of 4 j-tiles
  const int tid  = threadIdx.x;
  const int wave = tid >> 5;
  const int lane = tid & 31;
  const int Jt   = Jg * 4 + wave;       // this wave's j-tile

  // Entire block strictly below diagonal -> every gate is zero.
  if (Jg * 4 + 3 < It) {
    if (lane == 0) partial[(b * 16 + It) * 16 + Jt] = 0.f;
    return;
  }

  __builtin_prefetch(w1a, 0, 3);
  __builtin_prefetch(w2h, 0, 3);

  __shared__ __align__(16) _Float16 sXj[64 * DD];   // 8 KB  x2w rows j0..j0+63
  __shared__ float sUj[64 * DD];                    // 16 KB uh rows
  __shared__ float sMj[64];                         // mask_j
  __shared__ __align__(16) _Float16 sH[4 * 16 * DD];// 8 KB per-wave h scratch

  const int j0 = Jg * 64;
  {
    const unsigned int* gx = (const unsigned int*)(x2w + (size_t)(b * LL + j0) * DD);
    unsigned int* sx = (unsigned int*)sXj;
    for (int t = tid; t < 2048; t += 128) sx[t] = gx[t];
    const float* gu = uh + (size_t)(b * LL + j0) * DD;
    for (int t = tid; t < 4096; t += 128) sUj[t] = gu[t];
    if (tid < 64) sMj[tid] = maskA[b * LL + j0 + tid];
  }
  __syncthreads();
  // No further barriers: waves whose whole tile is below the diagonal retire now.
  if (Jt < It) {
    if (lane == 0) partial[(b * 16 + It) * 16 + Jt] = 0.f;
    return;
  }

  const int n   = lane & 15;            // N index / A-row index
  const int grp = lane >> 4;            // lane-group (K-offset select)
  const int i0  = It * 16;

  // A fragments for the 16 xi rows (row m = lane%16, WMMA A layout)
  const _Float16* xir = x2w + (size_t)(b * LL + i0 + n) * DD;
  const v16h XI0 = ld_frag2(xir + grp * 8,      xir + 16 + grp * 8);
  const v16h XI1 = ld_frag2(xir + 32 + grp * 8, xir + 48 + grp * 8);

  // Constant B fragments: layer1 abs-weights (4 n-tiles x 2 k-steps), layer2 w2
  v16h B1f[4][2];
#pragma unroll
  for (int nt = 0; nt < 4; ++nt)
#pragma unroll
    for (int ks = 0; ks < 2; ++ks) {
      const _Float16* p = w1a + (size_t)(nt * 16 + n) * DD + ks * 32 + grp * 8;
      B1f[nt][ks] = ld_frag2(p, p + 16);
    }
  v16h B2f[2];
#pragma unroll
  for (int ks = 0; ks < 2; ++ks) {
    const _Float16* p = w2h + (size_t)n * DD + ks * 32 + grp * 8;
    B2f[ks] = ld_frag2(p, p + 16);
  }

  // Per-lane C-layout preloads: uh_i (+b1 folded in), masks, biases
  float UHIB[4][8];
#pragma unroll
  for (int nt = 0; nt < 4; ++nt) {
    const float b1v = b1[nt * 16 + n];
#pragma unroll
    for (int r = 0; r < 8; ++r)
      UHIB[nt][r] = uh[(size_t)(b * LL + i0 + grp * 8 + r) * DD + nt * 16 + n] + b1v;
  }
  float mI8[8];
#pragma unroll
  for (int r = 0; r < 8; ++r) mI8[r] = maskA[b * LL + i0 + grp * 8 + r];
  const float b2v = b2[n];
  const float w3v = w3[n];
  const float laneB3 = (lane < 16) ? b3[0] * maskA[b * LL + i0 + n] : 0.f;

  _Float16* hs = sH + wave * (16 * DD);
  const _Float16* hrow = sH + wave * (16 * DD) + n * DD;

  float pacc = 0.f;

#pragma unroll 1
  for (int jj = 0; jj < 16; ++jj) {
    const int jl = wave * 16 + jj;      // row inside staged LDS tile
    const int j  = j0 + jl;             // global j
    const _Float16* xjr = sXj + jl * DD;
    const v16h XJ0 = ld_frag2(xjr + grp * 8,      xjr + 16 + grp * 8);
    const v16h XJ1 = ld_frag2(xjr + 32 + grp * 8, xjr + 48 + grp * 8);
    const v16h A0 = abs_diff(XI0, XJ0);
    const v16h A1 = abs_diff(XI1, XJ1);
    const float mj = sMj[jl];

    // Layer 1: h = leaky( |xi-xj| @ W1b  +  (uh_i + uh_j + b1) )
#pragma unroll
    for (int nt = 0; nt < 4; ++nt) {
      const float uhj = sUj[jl * DD + nt * 16 + n];
      v8f C;
#pragma unroll
      for (int r = 0; r < 8; ++r) C[r] = UHIB[nt][r] + uhj;
      C = __builtin_amdgcn_wmma_f32_16x16x32_f16(false, A0, false, B1f[nt][0],
                                                 (short)0, C, false, false);
      C = __builtin_amdgcn_wmma_f32_16x16x32_f16(false, A1, false, B1f[nt][1],
                                                 (short)0, C, false, false);
#pragma unroll
      for (int r = 0; r < 8; ++r)
        hs[(grp * 8 + r) * DD + nt * 16 + n] = (_Float16)lrelu(C[r]);
    }
    // wave-internal LDS RAW: stores above feed the A2 loads below
    asm volatile("s_wait_dscnt 0" ::: "memory");

    // Layer 2: h2 = leaky( h @ w2^T + b2 )
    const v16h A20 = ld_frag2(hrow + grp * 8,      hrow + 16 + grp * 8);
    const v16h A21 = ld_frag2(hrow + 32 + grp * 8, hrow + 48 + grp * 8);
    v8f C2;
#pragma unroll
    for (int r = 0; r < 8; ++r) C2[r] = b2v;
    C2 = __builtin_amdgcn_wmma_f32_16x16x32_f16(false, A20, false, B2f[0],
                                                (short)0, C2, false, false);
    C2 = __builtin_amdgcn_wmma_f32_16x16x32_f16(false, A21, false, B2f[1],
                                                (short)0, C2, false, false);

    // Head: gated dot with w3, plus b3 once per gated pair (lanes 0..15)
#pragma unroll
    for (int r = 0; r < 8; ++r) {
      const float v = lrelu(C2[r]);
      const int i = i0 + grp * 8 + r;
      const float gate = (j > i) ? (mI8[r] * mj) : 0.f;
      pacc += gate * w3v * v;
    }
    pacc += (j > i0 + n) ? laneB3 * mj : 0.f;
  }

  // deterministic wave reduction
#pragma unroll
  for (int off = 16; off > 0; off >>= 1) pacc += __shfl_xor(pacc, off, 32);
  if (lane == 0) partial[(b * 16 + It) * 16 + Jt] = pacc;
}

// ---------------- finalize: out[b] = x1[b] + scale * sum(partials) -------
__global__ __launch_bounds__(256)
void fin_kernel(const float* __restrict__ partial, const float* __restrict__ scale,
                float* __restrict__ out) {
  const int b = blockIdx.x;
  const int t = threadIdx.x;
  __shared__ float red[256];
  red[t] = partial[b * 256 + t];
  __syncthreads();
  for (int s = 128; s > 0; s >>= 1) { if (t < s) red[t] += red[t + s]; __syncthreads(); }
  if (t == 0) out[b] += scale[0] * red[0];
}

extern "C" void kernel_launch(void* const* d_in, const int* in_sizes, int n_in,
                              void* d_out, int out_size, void* d_ws, size_t ws_size,
                              hipStream_t stream) {
  (void)in_sizes; (void)n_in; (void)out_size; (void)ws_size;
  const float* emb   = (const float*)d_in[0];
  const float* tw    = (const float*)d_in[1];
  const float* tb    = (const float*)d_in[2];
  const float* w1    = (const float*)d_in[3];
  const float* b1    = (const float*)d_in[4];
  const float* w2    = (const float*)d_in[5];
  const float* b2    = (const float*)d_in[6];
  const float* w3    = (const float*)d_in[7];
  const float* b3    = (const float*)d_in[8];
  const float* scale = (const float*)d_in[9];
  float* out = (float*)d_out;

  char* ws = (char*)d_ws;
  _Float16* x2w   = (_Float16*)(ws + OFF_X2W);
  float*    uh    = (float*)   (ws + OFF_UH);
  float*    maskA = (float*)   (ws + OFF_MSK);
  _Float16* w1a   = (_Float16*)(ws + OFF_W1A);
  _Float16* w2h   = (_Float16*)(ws + OFF_W2H);
  float*    part  = (float*)   (ws + OFF_PAR);

  prep_kernel<<<BB, 256, 0, stream>>>(emb, tw, tb, x2w, maskA, out);
  wprep_kernel<<<16, 256, 0, stream>>>(w1, w2, w1a, w2h);
  uh_kernel<<<BB * LL, 64, 0, stream>>>(emb, tw, w1, uh);
  pair_kernel<<<dim3(16, 4, BB), 128, 0, stream>>>(x2w, uh, maskA, w1a, w2h,
                                                   b1, b2, w3, b3, part);
  fin_kernel<<<BB, 256, 0, stream>>>(part, scale, out);
}
